// CosineSimCodebook_56289841382017
// MI455X (gfx1250) — matrix operand order repaired and checked by reading, not compiled, not run on hardware
//
#include <hip/hip_runtime.h>
#include <hip/hip_bf16.h>

// ---------------------------------------------------------------------------
// CosineSimCodebook (VQ w/ EMA) for gfx1250.
//   x:     [8,4096,256] f32  -> N=32768 rows, d=256
//   embed: [1024,256]   f32  -> K=1024 codes
// Pipeline:
//   1) normalize embed rows -> en_f32 (ws) + en_bf16 (ws)
//   2) normalize x rows     -> xn_bf16 (ws)
//   3) WMMA bf16 GEMM [N,256]x[256,1024] with fused per-row argmax -> ind (ws)
//   4) gather quantize rows from raw embed -> d_out; scatter normalized rows
//      into embed_sum + bincount via f32 atomics
//   5) EMA finalize -> embed_new in d_out
// ---------------------------------------------------------------------------

typedef __bf16 bf16;
typedef __attribute__((ext_vector_type(16))) __bf16 v16bf;
typedef __attribute__((ext_vector_type(8)))  float  v8f;

#define D 256          // feature dim (fixed by reference)
#define KCODES 1024    // codebook size
#define BLK 256        // threads per block (8 waves, wave32)

__device__ __forceinline__ float block_reduce_sum256(float v, float* sm) {
    const int t = threadIdx.x;
    sm[t] = v;
    __syncthreads();
    #pragma unroll
    for (int s = 128; s > 0; s >>= 1) {
        if (t < s) sm[t] += sm[t + s];
        __syncthreads();
    }
    float r = sm[0];
    __syncthreads();
    return r;
}

// -------------------- 0) zero the EMA accumulators -------------------------
__global__ __launch_bounds__(BLK) void vq_init_stats(float* __restrict__ p, int n) {
    int i = blockIdx.x * BLK + threadIdx.x;
    if (i < n) p[i] = 0.0f;
}

// -------------------- 1) normalize codebook rows ---------------------------
__global__ __launch_bounds__(BLK) void vq_norm_embed(const float* __restrict__ embed,
                                                     float* __restrict__ en_f32,
                                                     bf16*  __restrict__ en_bf16) {
    __shared__ float sm[BLK];
    const int k = blockIdx.x;
    const int t = threadIdx.x;
    const float v = embed[(size_t)k * D + t];
    const float ss = block_reduce_sum256(v * v, sm);
    const float inv = 1.0f / fmaxf(sqrtf(ss), 1e-12f);
    const float nv = v * inv;
    en_f32[(size_t)k * D + t] = nv;
    en_bf16[(size_t)k * D + t] = (bf16)nv;
}

// -------------------- 2) normalize x rows ----------------------------------
__global__ __launch_bounds__(BLK) void vq_norm_x(const float* __restrict__ x,
                                                 bf16* __restrict__ xn_bf16) {
    __shared__ float sm[BLK];
    const size_t row = blockIdx.x;
    const int t = threadIdx.x;
    const float v = x[row * D + t];
    const float ss = block_reduce_sum256(v * v, sm);
    const float inv = 1.0f / fmaxf(sqrtf(ss), 1e-12f);
    xn_bf16[row * D + t] = (bf16)(v * inv);
}

// -------------------- 3) WMMA GEMM + fused argmax --------------------------
// One wave computes a 16-row tile vs all 1024 codes.
// A fragment layout (16-bit A 16x32, ISA 7.12.2): lane (0..15 | 16..31) holds
// row (lane&15); K chunks [hi*8 .. hi*8+7] and [hi*8+16 .. hi*8+23], hi=lane>>4.
// B fragment layout (dense analog of 16-bit B): lane&15 = column N,
// lane>>4 selects K [hi*8 .. hi*8+7] and [hi*8+16 .. hi*8+23], same order as A.
__global__ __launch_bounds__(BLK) void vq_gemm_argmax(const bf16* __restrict__ xn,
                                                      const bf16* __restrict__ en,
                                                      int* __restrict__ ind) {
    const int lane   = threadIdx.x & 31;
    const int wave   = threadIdx.x >> 5;
    const int laneLo = lane & 15;
    const int hi     = lane >> 4;           // 0 or 1
    const int rowTile = blockIdx.x * 8 + wave;
    const int rowBase = rowTile * 16;

    // ---- preload all 8 A fragments (K = 256 = 8 steps of 32) into VGPRs ----
    const bf16* arow = xn + (size_t)(rowBase + laneLo) * D;
    v16bf afrag[8];
    #pragma unroll
    for (int ks = 0; ks < 8; ++ks) {
        const bf16* p = arow + ks * 32 + hi * 8;
        #pragma unroll
        for (int i = 0; i < 8; ++i) {
            afrag[ks][i]     = p[i];        // K = hi*8 + 0..7
            afrag[ks][8 + i] = p[16 + i];   // K = hi*8 + 16..23
        }
    }

    float bestVal[8];
    int   bestIdx[8];
    #pragma unroll
    for (int r = 0; r < 8; ++r) { bestVal[r] = -3.0e38f; bestIdx[r] = 0; }

    for (int ct = 0; ct < KCODES / 16; ++ct) {
        const int colBase = ct * 16;
        const bf16* brow = en + (size_t)(colBase + laneLo) * D + hi * 16;
        if (ct + 1 < KCODES / 16) {
            // warm L2/WGP$ for next column tile (global_prefetch_b8)
            __builtin_prefetch(en + (size_t)(colBase + 16 + laneLo) * D, 0, 1);
        }
        v8f acc = {};
        #pragma unroll
        for (int ks = 0; ks < 8; ++ks) {
            const bf16* p = brow + ks * 32;
            v16bf bfrag;
            #pragma unroll
            for (int i = 0; i < 16; ++i) bfrag[i] = p[(i & 8) * 2 - hi * 8 + (i & 7)];
            acc = __builtin_amdgcn_wmma_f32_16x16x32_bf16(
                      /*neg_a=*/false, afrag[ks],
                      /*neg_b=*/false, bfrag,
                      /*c_mod=*/(short)0, acc,
                      /*reuse_a=*/false, /*reuse_b=*/false);
        }
        // acc[r]: score for row rowBase + hi*8 + r, col colBase + laneLo
        const int c = colBase + laneLo;
        #pragma unroll
        for (int r = 0; r < 8; ++r) {
            const float v = acc[r];
            if (v > bestVal[r]) { bestVal[r] = v; bestIdx[r] = c; }
        }
    }

    // reduce argmax across the 16 lanes of each half (xor 1/2/4/8 keeps halves)
    #pragma unroll
    for (int m = 1; m <= 8; m <<= 1) {
        #pragma unroll
        for (int r = 0; r < 8; ++r) {
            const float ov = __shfl_xor(bestVal[r], m, 32);
            const int   oi = __shfl_xor(bestIdx[r], m, 32);
            if (ov > bestVal[r] || (ov == bestVal[r] && oi < bestIdx[r])) {
                bestVal[r] = ov; bestIdx[r] = oi;
            }
        }
    }
    if (laneLo == 0) {
        #pragma unroll
        for (int r = 0; r < 8; ++r)
            ind[rowBase + hi * 8 + r] = bestIdx[r];
    }
}

// -------------------- 4) gather quantize + scatter EMA stats ---------------
__global__ __launch_bounds__(BLK) void vq_gather_scatter(const float* __restrict__ x,
                                                         const float* __restrict__ embed,
                                                         const int* __restrict__ ind,
                                                         float* __restrict__ out_q,
                                                         float* __restrict__ out_ind,
                                                         float* __restrict__ bins,
                                                         float* __restrict__ esum) {
    __shared__ float sm[BLK];
    const size_t row = blockIdx.x;
    const int t = threadIdx.x;
    const int k = ind[row];

    // quantize = raw embed gather
    out_q[row * D + t] = embed[(size_t)k * D + t];
    if (t == 0) {
        out_ind[row] = (float)k;
        atomicAdd(&bins[k], 1.0f);
    }
    // segment-sum of the l2-normalized row (recomputed in f32)
    const float v = x[row * D + t];
    const float ss = block_reduce_sum256(v * v, sm);
    const float inv = 1.0f / fmaxf(sqrtf(ss), 1e-12f);
    atomicAdd(&esum[(size_t)k * D + t], v * inv);
}

// -------------------- 5) EMA finalize --------------------------------------
__global__ __launch_bounds__(BLK) void vq_ema(const float* __restrict__ embed,
                                              const float* __restrict__ en_f32,
                                              const float* __restrict__ bins,
                                              const float* __restrict__ esum,
                                              float* __restrict__ out_e) {
    __shared__ float sm[BLK];
    const int k = blockIdx.x;
    const int t = threadIdx.x;
    const float b = bins[k];
    const int dead = (b == 0.0f);
    const float mean = dead ? 0.0f : esum[(size_t)k * D + t] / b;
    const float ss = block_reduce_sum256(mean * mean, sm);
    float normed;
    if (dead) normed = en_f32[(size_t)k * D + t];
    else      normed = mean / fmaxf(sqrtf(ss), 1e-12f);
    out_e[(size_t)k * D + t] = embed[(size_t)k * D + t] * 0.8f + normed * 0.2f;
}

// ---------------------------------------------------------------------------
extern "C" void kernel_launch(void* const* d_in, const int* in_sizes, int n_in,
                              void* d_out, int out_size, void* d_ws, size_t ws_size,
                              hipStream_t stream) {
    const float* x     = (const float*)d_in[0];
    const float* embed = (const float*)d_in[1];

    const int N = in_sizes[0] / D;      // 32768 rows

    // workspace layout (all 16B-aligned)
    char* ws = (char*)d_ws;
    bf16*  xn_bf16 = (bf16*)ws;                         size_t o = (size_t)N * D * sizeof(bf16);
    o = (o + 255) & ~(size_t)255;
    bf16*  en_bf16 = (bf16*)(ws + o);                   o += (size_t)KCODES * D * sizeof(bf16);
    o = (o + 255) & ~(size_t)255;
    float* en_f32  = (float*)(ws + o);                  o += (size_t)KCODES * D * sizeof(float);
    o = (o + 255) & ~(size_t)255;
    int*   ind     = (int*)(ws + o);                    o += (size_t)N * sizeof(int);
    o = (o + 255) & ~(size_t)255;
    float* bins    = (float*)(ws + o);                  o += (size_t)KCODES * sizeof(float);
    o = (o + 255) & ~(size_t)255;
    float* esum    = (float*)(ws + o);

    // output layout: [quantize (N*D) | embed_ind (N) | embed_new (K*D)]
    float* out_q   = (float*)d_out;
    float* out_ind = out_q + (size_t)N * D;
    float* out_e   = out_ind + N;

    const int statsN = KCODES + KCODES * D;
    vq_init_stats<<<(statsN + BLK - 1) / BLK, BLK, 0, stream>>>(bins, statsN);  // bins+esum contiguous
    vq_norm_embed<<<KCODES, BLK, 0, stream>>>(embed, en_f32, en_bf16);
    vq_norm_x<<<N, BLK, 0, stream>>>(x, xn_bf16);
    vq_gemm_argmax<<<N / 128, BLK, 0, stream>>>(xn_bf16, en_bf16, ind);         // 8 waves x 16 rows
    vq_gather_scatter<<<N, BLK, 0, stream>>>(x, embed, ind, out_q, out_ind, bins, esum);
    vq_ema<<<KCODES, BLK, 0, stream>>>(embed, en_f32, bins, esum, out_e);
}
